// SwinDecoder_23441931502141
// MI455X (gfx1250) — matrix operand order repaired and checked by reading, not compiled, not run on hardware
//
#include <hip/hip_runtime.h>
#include <hip/hip_bf16.h>
#include <math.h>

// ---------------------------------------------------------------------------
// Types for CDNA5 WMMA (gfx1250, wave32)
// ---------------------------------------------------------------------------
typedef __bf16 bf16;
typedef bf16  v16bf __attribute__((ext_vector_type(16)));
typedef bf16  v8bf  __attribute__((ext_vector_type(8)));
typedef float v8f   __attribute__((ext_vector_type(8)));

#define WMMA_BF16(A, B, C) \
    __builtin_amdgcn_wmma_f32_16x16x32_bf16(false, (A), false, (B), (short)0, (C), false, false)

// Model constants
#define TTOK 65536      // B(2) * 32^3 tokens
#define CCH  192
#define HEADS 6
#define HD   32
#define NTOK 128        // tokens per window (2*8*8)

// ---------------------------------------------------------------------------
// CDNA5 async global->LDS copy (ASYNCcnt-tracked, no VGPR round trip)
// ---------------------------------------------------------------------------
typedef __attribute__((address_space(3))) void lds_void;

__device__ __forceinline__ unsigned lds_addr32(const void* generic) {
    return (unsigned)(unsigned long long)(lds_void*)generic;
}
__device__ __forceinline__ void async_load_b128(unsigned lds_off, const void* gptr) {
    asm volatile("global_load_async_to_lds_b128 %0, %1, off"
                 :: "v"(lds_off), "v"(gptr)
                 : "memory");
}
__device__ __forceinline__ void async_wait0() {
    asm volatile("s_wait_asynccnt 0" ::: "memory");
}

// ---------------------------------------------------------------------------
// Generic bf16 WMMA GEMM: Out[M,N] = A[M,K] @ W^T (W stored N x K) + epilogue
// Block tile 128x64, 8 waves, each wave a 32x32 subtile (2x2 WMMAs).
// K stepped by 32.  Double-buffered LDS staged with async loads; the
// software pipeline is unrolled 2x so buffer indices are static (requires an
// even number of K-steps, which holds for K in {128,192,768}).
// ---------------------------------------------------------------------------
#define GT_M 128
#define GT_N 64
#define LDA_S 40   // LDS row stride (bf16) for A tile (32 + 8 pad)
#define LDB_S 40

__global__ __launch_bounds__(256) void gemm_bf16_wmma(
    const bf16* __restrict__ A,      // M x K row-major
    const bf16* __restrict__ W,      // N x K row-major (pre-transposed weight)
    const float* __restrict__ bias,  // len N or nullptr
    const float* __restrict__ resid, // M x N or nullptr
    float* __restrict__ outf,        // M x N or nullptr
    bf16* __restrict__ outb,         // M x N or nullptr
    int M, int N, int K, int gelu)
{
    __shared__ __align__(16) bf16 As[2][GT_M * LDA_S];
    __shared__ __align__(16) bf16 Bs[2][GT_N * LDB_S];

    const int m0   = blockIdx.x * GT_M;
    const int n0   = blockIdx.y * GT_N;
    const int tid  = threadIdx.x;
    const int wave = tid >> 5;
    const int lane = tid & 31;
    const int wm   = wave & 3;    // 4 waves along M (32 rows each)
    const int wn   = wave >> 2;   // 2 waves along N (32 cols each)
    const int half = lane >> 4;
    const int l16  = lane & 15;

    // Per-thread staging coordinates
    const int arow = tid >> 1, apart = tid & 1;        // A: 2 thr/row, 32B each
    const int brow = tid >> 2, bpart = tid & 3;        // W: 4 thr/row, 16B each
    const unsigned aoffA[2] = {
        lds_addr32(&As[0][arow * LDA_S + apart * 16]),
        lds_addr32(&As[1][arow * LDA_S + apart * 16]) };
    const unsigned boffB[2] = {
        lds_addr32(&Bs[0][brow * LDB_S + bpart * 8]),
        lds_addr32(&Bs[1][brow * LDB_S + bpart * 8]) };
    const bf16* aSrcRow = A + (size_t)(m0 + arow) * K + apart * 16;
    const bf16* bSrcRow = W + (size_t)(n0 + brow) * K + bpart * 8;

    v8f acc[2][2];
#pragma unroll
    for (int i = 0; i < 2; ++i)
#pragma unroll
        for (int j = 0; j < 2; ++j)
#pragma unroll
            for (int r = 0; r < 8; ++r) acc[i][j][r] = 0.0f;

    // Stage K-step `kstep` into LDS buffer `sbuf` (3 async b128 per thread)
    auto stage = [&](int sbuf, int kstep) {
        const bf16* aS = aSrcRow + kstep * 32;
        const bf16* bS = bSrcRow + kstep * 32;
        async_load_b128(aoffA[sbuf],      aS);
        async_load_b128(aoffA[sbuf] + 16, aS + 8);
        async_load_b128(boffB[sbuf],      bS);
    };
    // Consume LDS buffer `cbuf`: load fragments once, run the 2x2 WMMA quad
    auto compute = [&](int cbuf) {
        const bf16* Ab = As[cbuf];
        const bf16* Bb = Bs[cbuf];
        v16bf af[2], bfr[2];
#pragma unroll
        for (int i = 0; i < 2; ++i) {
            int am = wm * 32 + i * 16 + l16;
            v8bf a0 = *(const v8bf*)(&Ab[am * LDA_S + half * 8]);
            v8bf a1 = *(const v8bf*)(&Ab[am * LDA_S + 16 + half * 8]);
            af[i] = __builtin_shufflevector(a0, a1,
                0, 1, 2, 3, 4, 5, 6, 7, 8, 9, 10, 11, 12, 13, 14, 15);
        }
#pragma unroll
        for (int j = 0; j < 2; ++j) {
            int bn = wn * 32 + j * 16 + l16;
            v8bf b0 = *(const v8bf*)(&Bb[bn * LDB_S + half * 8]);
            v8bf b1 = *(const v8bf*)(&Bb[bn * LDB_S + 16 + half * 8]);
            bfr[j] = __builtin_shufflevector(b0, b1,
                0, 1, 2, 3, 4, 5, 6, 7, 8, 9, 10, 11, 12, 13, 14, 15);
        }
#pragma unroll
        for (int i = 0; i < 2; ++i)
#pragma unroll
            for (int j = 0; j < 2; ++j)
                acc[i][j] = WMMA_BF16(af[i], bfr[j], acc[i][j]);
    };

    const int nk = K >> 5;   // even for all K used here
    stage(0, 0);
    async_wait0();
    __syncthreads();

    for (int it = 0; it < nk; it += 2) {
        // Phase A: stage buf1 (k-step it+1) while computing buf0 (k-step it)
        stage(1, it + 1);
        compute(0);
        async_wait0();
        __syncthreads();
        // Phase B: stage buf0 (k-step it+2) while computing buf1 (k-step it+1)
        if (it + 2 < nk) stage(0, it + 2);
        compute(1);
        async_wait0();
        __syncthreads();
    }

    // Epilogue
#pragma unroll
    for (int i = 0; i < 2; ++i)
#pragma unroll
        for (int j = 0; j < 2; ++j) {
            int gn = n0 + wn * 32 + j * 16 + l16;
#pragma unroll
            for (int r = 0; r < 8; ++r) {
                int gm = m0 + wm * 32 + i * 16 + half * 8 + r;
                float v = acc[i][j][r];
                if (bias)  v += bias[gn];
                if (gelu)  v = 0.5f * v * (1.0f + erff(v * 0.70710678118654752f));
                if (resid) v += resid[(size_t)gm * N + gn];
                if (outf)  outf[(size_t)gm * N + gn] = v;
                if (outb)  outb[(size_t)gm * N + gn] = (bf16)v;
            }
        }
}

// ---------------------------------------------------------------------------
// Fused window attention.  One block per (window, head).  256 threads.
//   S = (Q*scale) @ K^T  (128x128, K=32, WMMA)
//   S += rel-pos bias (+ shift mask), softmax rows
//   O = P @ V            (128x32,  K=128, WMMA)
// ---------------------------------------------------------------------------
__device__ __forceinline__ int rp_index(int n, int m) {
    int dz = (n >> 6) - (m >> 6) + 1;          // (2*WZ-1)=3
    int dh = ((n >> 3) & 7) - ((m >> 3) & 7) + 7;
    int dw = (n & 7) - (m & 7) + 7;
    return dz * 225 + dh * 15 + dw;            // 225 = 15*15
}
__device__ __forceinline__ int regionof(int p, int P, int Wd, int s) {
    return (p < P - Wd) ? 0 : ((p < P - s) ? 1 : 2);
}

#define SM_V   0            // bf16 Vs[32][136]   -> 8704 B
#define SM_Q   8704         // bf16 Qs[128][40]   -> 10240 B
#define SM_K   18944        // bf16 Ks[128][40]   -> 10240 B
#define SM_P   8704         // bf16 Ps[128][136]  -> 34816 B (aliases Q/K)
#define SM_B   43520        // float bias[675]    -> 2700 B
#define SM_L   46224        // int labels[128]    -> 512 B
#define SM_SZ  46848

__global__ __launch_bounds__(256) void window_attn_kernel(
    const float* __restrict__ qkv,   // [T][576], window-token order
    const float* __restrict__ rpb_l, // [675][6] for this layer
    bf16* __restrict__ attn_out,     // [T][192] bf16, window-token order
    int shifted)
{
    __shared__ __align__(16) char smem[SM_SZ];
    bf16*  Vs    = (bf16*)(smem + SM_V);
    bf16*  Qs    = (bf16*)(smem + SM_Q);
    bf16*  Ks    = (bf16*)(smem + SM_K);
    bf16*  Ps    = (bf16*)(smem + SM_P);
    float* biasS = (float*)(smem + SM_B);
    int*   labS  = (int*)(smem + SM_L);

    const int bw   = blockIdx.x;       // b*NWIN + win (0..511)
    const int head = blockIdx.y;       // 0..5
    const int t0   = bw * NTOK;
    const int tid  = threadIdx.x;
    const int wave = tid >> 5;
    const int lane = tid & 31;
    const int half = lane >> 4;
    const int l16  = lane & 15;
    const float scale = 0.17677669529663688f;   // 32^-0.5

    // ---- Stage Q (scaled), K, V(transposed) into LDS as bf16 ----
    for (int i = tid; i < NTOK * HD; i += 256) {
        int row = i >> 5, c = i & 31;
        const float* base = qkv + (size_t)(t0 + row) * 576 + head * HD + c;
        Qs[row * 40 + c]  = (bf16)(base[0] * scale);
        Ks[row * 40 + c]  = (bf16)(base[192]);
        Vs[c * 136 + row] = (bf16)(base[384]);
    }
    for (int i = tid; i < 675; i += 256) biasS[i] = rpb_l[i * 6 + head];
    if (tid < NTOK) {
        int lab = 0;
        if (shifted) {
            int win = bw & 255;
            int wz = win >> 4, wh = (win >> 2) & 3, ww = win & 3;
            int lz = tid >> 6, lh = (tid >> 3) & 7, lw = tid & 7;
            int rz = regionof(wz * 2 + lz, 32, 2, 1);
            int rh = regionof(wh * 8 + lh, 32, 8, 4);
            int rw = regionof(ww * 8 + lw, 32, 8, 4);
            lab = rz * 9 + rh * 3 + rw;
        }
        labS[tid] = lab;
    }
    __syncthreads();

    // ---- S = Q @ K^T : wave owns rows [wave*16, wave*16+16) ----
    v8f s[8];
    {
        int am = wave * 16 + l16;
        v8bf a0 = *(const v8bf*)(&Qs[am * 40 + half * 8]);
        v8bf a1 = *(const v8bf*)(&Qs[am * 40 + 16 + half * 8]);
        v16bf qfrag = __builtin_shufflevector(a0, a1,
            0, 1, 2, 3, 4, 5, 6, 7, 8, 9, 10, 11, 12, 13, 14, 15);
#pragma unroll
        for (int t = 0; t < 8; ++t) {
            int bn = t * 16 + l16;
            v8bf b0 = *(const v8bf*)(&Ks[bn * 40 + half * 8]);
            v8bf b1 = *(const v8bf*)(&Ks[bn * 40 + 16 + half * 8]);
            v16bf kfrag = __builtin_shufflevector(b0, b1,
                0, 1, 2, 3, 4, 5, 6, 7, 8, 9, 10, 11, 12, 13, 14, 15);
            v8f z;
#pragma unroll
            for (int r = 0; r < 8; ++r) z[r] = 0.0f;
            s[t] = WMMA_BF16(qfrag, kfrag, z);
        }
    }
    __syncthreads();   // all waves done reading Qs/Ks; safe to alias via Ps

    // ---- bias + mask + softmax, write P (bf16) ----
#pragma unroll
    for (int r = 0; r < 8; ++r) {
        int nrow = wave * 16 + half * 8 + r;
        int labn = labS[nrow];
        float mx = -1e30f;
#pragma unroll
        for (int t = 0; t < 8; ++t) {
            int mcol = t * 16 + l16;
            float v = s[t][r] + biasS[rp_index(nrow, mcol)];
            if (shifted && (labn != labS[mcol])) v -= 100.0f;
            s[t][r] = v;
            mx = fmaxf(mx, v);
        }
        for (int o = 1; o < 16; o <<= 1) mx = fmaxf(mx, __shfl_xor(mx, o, 32));
        float sum = 0.0f;
#pragma unroll
        for (int t = 0; t < 8; ++t) {
            float p = __expf(s[t][r] - mx);
            s[t][r] = p;
            sum += p;
        }
        for (int o = 1; o < 16; o <<= 1) sum += __shfl_xor(sum, o, 32);
        float inv = 1.0f / sum;
#pragma unroll
        for (int t = 0; t < 8; ++t)
            Ps[nrow * 136 + t * 16 + l16] = (bf16)(s[t][r] * inv);
    }
    __syncthreads();

    // ---- O = P @ V : 128x32, K=128 in 4 steps ----
    v8f oacc[2];
#pragma unroll
    for (int j = 0; j < 2; ++j)
#pragma unroll
        for (int r = 0; r < 8; ++r) oacc[j][r] = 0.0f;

#pragma unroll
    for (int kk = 0; kk < 4; ++kk) {
        int k0 = kk * 32;
        int am = wave * 16 + l16;
        v8bf a0 = *(const v8bf*)(&Ps[am * 136 + k0 + half * 8]);
        v8bf a1 = *(const v8bf*)(&Ps[am * 136 + k0 + 16 + half * 8]);
        v16bf pfrag = __builtin_shufflevector(a0, a1,
            0, 1, 2, 3, 4, 5, 6, 7, 8, 9, 10, 11, 12, 13, 14, 15);
#pragma unroll
        for (int j = 0; j < 2; ++j) {
            int bn = j * 16 + l16;
            v8bf b0 = *(const v8bf*)(&Vs[bn * 136 + k0 + half * 8]);
            v8bf b1 = *(const v8bf*)(&Vs[bn * 136 + k0 + 16 + half * 8]);
            v16bf vfrag = __builtin_shufflevector(b0, b1,
                0, 1, 2, 3, 4, 5, 6, 7, 8, 9, 10, 11, 12, 13, 14, 15);
            oacc[j] = WMMA_BF16(pfrag, vfrag, oacc[j]);
        }
    }

#pragma unroll
    for (int j = 0; j < 2; ++j)
#pragma unroll
        for (int r = 0; r < 8; ++r) {
            int token = t0 + wave * 16 + half * 8 + r;
            int ch    = head * HD + j * 16 + l16;
            attn_out[(size_t)token * CCH + ch] = (bf16)oacc[j][r];
        }
}

// ---------------------------------------------------------------------------
// LayerNorm over C=192.  One wave per token (8 tokens / 256-thread block).
// mode 0: identity.  mode 1: window partition.  mode 2: +shift.
// ---------------------------------------------------------------------------
__global__ __launch_bounds__(256) void ln_kernel(
    const float* __restrict__ in,   // [T][192] spatial order
    const float* __restrict__ g, const float* __restrict__ b,
    float* __restrict__ outf, bf16* __restrict__ outb, int mode)
{
    const int token = blockIdx.x * 8 + (threadIdx.x >> 5);  // output index
    const int lane  = threadIdx.x & 31;

    int src = token;
    if (mode != 0) {
        int bw = token >> 7, n = token & 127;
        int bb = bw >> 8, win = bw & 255;
        int wz = win >> 4, wh = (win >> 2) & 3, ww = win & 3;
        int lz = n >> 6, lh = (n >> 3) & 7, lw = n & 7;
        int z = wz * 2 + lz, h = wh * 8 + lh, w = ww * 8 + lw;
        if (mode == 2) { z = (z + 1) & 31; h = (h + 4) & 31; w = (w + 4) & 31; }
        src = ((bb * 32 + z) * 32 + h) * 32 + w;
    }

    const float* row = in + (size_t)src * CCH;
    float vals[6], s = 0.0f, s2 = 0.0f;
#pragma unroll
    for (int k = 0; k < 6; ++k) {
        float x = row[lane + 32 * k];
        vals[k] = x; s += x; s2 += x * x;
    }
    for (int o = 16; o; o >>= 1) { s += __shfl_xor(s, o, 32); s2 += __shfl_xor(s2, o, 32); }
    float mean = s * (1.0f / 192.0f);
    float var  = s2 * (1.0f / 192.0f) - mean * mean;
    float inv  = rsqrtf(var + 1e-5f);
#pragma unroll
    for (int k = 0; k < 6; ++k) {
        int c = lane + 32 * k;
        float y = (vals[k] - mean) * inv * g[c] + b[c];
        if (outf) outf[(size_t)token * CCH + c] = y;
        if (outb) outb[(size_t)token * CCH + c] = (bf16)y;
    }
}

// Un-partition (+unshift) attention-proj output and add to residual stream.
__global__ __launch_bounds__(256) void merge_kernel(
    float* __restrict__ e, const float* __restrict__ proj, int shifted)
{
    int idx = blockIdx.x * 256 + threadIdx.x;   // T*192
    int token = idx / CCH, c = idx % CCH;
    int bb = token >> 15, z = (token >> 10) & 31, h = (token >> 5) & 31, w = token & 31;
    if (shifted) { z = (z + 31) & 31; h = (h + 28) & 31; w = (w + 28) & 31; }
    int wz = z >> 1, lz = z & 1;
    int wh = h >> 3, lh = h & 7;
    int ww = w >> 3, lw = w & 7;
    int win = (wz * 4 + wh) * 4 + ww;
    int n   = (lz * 8 + lh) * 8 + lw;
    int wt  = (((bb << 8) + win) << 7) + n;
    e[(size_t)token * CCH + c] += proj[(size_t)wt * CCH + c];
}

// Patchify: gather x into bf16 A matrix [T][128]
__global__ __launch_bounds__(256) void gather_patch_kernel(
    const float* __restrict__ x, bf16* __restrict__ A)
{
    int idx = blockIdx.x * 256 + threadIdx.x;   // T*128
    int token = idx >> 7, kk = idx & 127;
    int bb = token >> 15, z = (token >> 10) & 31, h = (token >> 5) & 31, w = token & 31;
    int c = kk >> 5, i = (kk >> 4) & 1, j = (kk >> 2) & 3, k = kk & 3;
    size_t src = (((size_t)(bb * 4 + c) * 64 + (2 * z + i)) * 128 + (4 * h + j)) * 128 + (4 * w + k);
    A[idx] = (bf16)x[src];
}

// Reconstruction scatter: out[b,o,2z+i,4h+j,4w+k] = tmp[token][n] + rec_b[o]
__global__ __launch_bounds__(256) void scatter_rec_kernel(
    const float* __restrict__ tmp, const float* __restrict__ rec_b,
    float* __restrict__ out)
{
    int idx = blockIdx.x * 256 + threadIdx.x;   // T*128
    int token = idx >> 7, n = idx & 127;
    int bb = token >> 15, z = (token >> 10) & 31, h = (token >> 5) & 31, w = token & 31;
    int o = n >> 5, i = (n >> 4) & 1, j = (n >> 2) & 3, k = n & 3;
    size_t dst = (((size_t)(bb * 4 + o) * 64 + (2 * z + i)) * 128 + (4 * h + j)) * 128 + (4 * w + k);
    out[dst] = tmp[idx] + rec_b[o];
}

// f32 -> bf16 elementwise
__global__ __launch_bounds__(256) void convert_kernel(
    const float* __restrict__ in, bf16* __restrict__ out, int n)
{
    int idx = blockIdx.x * 256 + threadIdx.x;
    if (idx < n) out[idx] = (bf16)in[idx];
}

// f32 KxN -> bf16 NxK (weight transpose-convert)
__global__ __launch_bounds__(256) void transpose_conv_kernel(
    const float* __restrict__ in, bf16* __restrict__ out, int K, int N)
{
    int idx = blockIdx.x * 256 + threadIdx.x;
    if (idx < K * N) {
        int k = idx / N, n = idx % N;
        out[(size_t)n * K + k] = (bf16)in[idx];
    }
}

// ---------------------------------------------------------------------------
// Host-side orchestration
// ---------------------------------------------------------------------------
extern "C" void kernel_launch(void* const* d_in, const int* in_sizes, int n_in,
                              void* d_out, int out_size, void* d_ws, size_t ws_size,
                              hipStream_t stream) {
    (void)in_sizes; (void)n_in; (void)out_size; (void)ws_size;
    const float* x       = (const float*)d_in[0];
    const float* pe_w    = (const float*)d_in[1];
    const float* pe_b    = (const float*)d_in[2];
    const float* pe_ln_g = (const float*)d_in[3];
    const float* pe_ln_b = (const float*)d_in[4];
    const float* ln1_g   = (const float*)d_in[5];
    const float* ln1_b   = (const float*)d_in[6];
    const float* qkv_w   = (const float*)d_in[7];
    const float* qkv_b   = (const float*)d_in[8];
    const float* proj_w  = (const float*)d_in[9];
    const float* proj_b  = (const float*)d_in[10];
    const float* rpb     = (const float*)d_in[11];
    const float* ln2_g   = (const float*)d_in[12];
    const float* ln2_b   = (const float*)d_in[13];
    const float* fc1_w   = (const float*)d_in[14];
    const float* fc1_b   = (const float*)d_in[15];
    const float* fc2_w   = (const float*)d_in[16];
    const float* fc2_b   = (const float*)d_in[17];
    const float* rec_w   = (const float*)d_in[18];
    const float* rec_b   = (const float*)d_in[19];

    const size_t T = TTOK;
    char* ws = (char*)d_ws;
    size_t off = 0;
    auto take = [&](size_t bytes) -> char* {
        char* p = ws + off;
        off += (bytes + 255) & ~(size_t)255;
        return p;
    };
    float* e      = (float*)take(T * 192 * 4);
    float* qkvbuf = (float*)take(T * 576 * 4);
    float* proj_t = (float*)take(T * 192 * 4);
    bf16*  A_w    = (bf16*)take(T * 192 * 2);
    bf16*  attn_o = (bf16*)take(T * 192 * 2);
    bf16*  hmid   = (bf16*)take(T * 768 * 2);
    bf16*  peW    = (bf16*)take(192 * 128 * 2);
    bf16*  qkvW   = (bf16*)take(4 * 576 * 192 * 2);
    bf16*  projW  = (bf16*)take(4 * 192 * 192 * 2);
    bf16*  fc1W   = (bf16*)take(4 * 768 * 192 * 2);
    bf16*  fc2W   = (bf16*)take(4 * 192 * 768 * 2);
    bf16*  recW   = (bf16*)take(128 * 192 * 2);
    // time-disjoint aliases
    bf16*  A_pe   = (bf16*)qkvbuf;   // used before any qkv GEMM
    float* rec_t  = (float*)qkvbuf;  // used after last block
    float* e_raw  = proj_t;          // used before first proj GEMM

    // ---- Weight conversions ----
    {
        int n = 192 * 128;   // pe_w is already [N=192][K=128]
        convert_kernel<<<(n + 255) / 256, 256, 0, stream>>>(pe_w, peW, n);
        transpose_conv_kernel<<<(192 * 128 + 255) / 256, 256, 0, stream>>>(rec_w, recW, 192, 128);
        for (int d = 0; d < 4; ++d) {
            transpose_conv_kernel<<<(192 * 576 + 255) / 256, 256, 0, stream>>>(
                qkv_w + (size_t)d * 192 * 576, qkvW + (size_t)d * 576 * 192, 192, 576);
            transpose_conv_kernel<<<(192 * 192 + 255) / 256, 256, 0, stream>>>(
                proj_w + (size_t)d * 192 * 192, projW + (size_t)d * 192 * 192, 192, 192);
            transpose_conv_kernel<<<(192 * 768 + 255) / 256, 256, 0, stream>>>(
                fc1_w + (size_t)d * 192 * 768, fc1W + (size_t)d * 768 * 192, 192, 768);
            transpose_conv_kernel<<<(768 * 192 + 255) / 256, 256, 0, stream>>>(
                fc2_w + (size_t)d * 768 * 192, fc2W + (size_t)d * 192 * 768, 768, 192);
        }
    }

    const int MB = TTOK / GT_M;   // 512 M-blocks

    // ---- Patch embed: e = LN(gather(x) @ peW + pe_b) ----
    gather_patch_kernel<<<(T * 128) / 256, 256, 0, stream>>>(x, A_pe);
    gemm_bf16_wmma<<<dim3(MB, 192 / GT_N), 256, 0, stream>>>(
        A_pe, peW, pe_b, nullptr, e_raw, nullptr, TTOK, 192, 128, 0);
    ln_kernel<<<TTOK / 8, 256, 0, stream>>>(e_raw, pe_ln_g, pe_ln_b, e, nullptr, 0);

    // ---- 4 Swin blocks ----
    for (int d = 0; d < 4; ++d) {
        int sh = d & 1;
        ln_kernel<<<TTOK / 8, 256, 0, stream>>>(
            e, ln1_g + d * 192, ln1_b + d * 192, nullptr, A_w, sh ? 2 : 1);
        gemm_bf16_wmma<<<dim3(MB, 576 / GT_N), 256, 0, stream>>>(
            A_w, qkvW + (size_t)d * 576 * 192, qkv_b + d * 576, nullptr,
            qkvbuf, nullptr, TTOK, 576, 192, 0);
        window_attn_kernel<<<dim3(512, HEADS), 256, 0, stream>>>(
            qkvbuf, rpb + (size_t)d * 675 * 6, attn_o, sh);
        gemm_bf16_wmma<<<dim3(MB, 192 / GT_N), 256, 0, stream>>>(
            attn_o, projW + (size_t)d * 192 * 192, proj_b + d * 192, nullptr,
            proj_t, nullptr, TTOK, 192, 192, 0);
        merge_kernel<<<(T * 192) / 256, 256, 0, stream>>>(e, proj_t, sh);
        ln_kernel<<<TTOK / 8, 256, 0, stream>>>(
            e, ln2_g + d * 192, ln2_b + d * 192, nullptr, A_w, 0);
        gemm_bf16_wmma<<<dim3(MB, 768 / GT_N), 256, 0, stream>>>(
            A_w, fc1W + (size_t)d * 768 * 192, fc1_b + d * 768, nullptr,
            nullptr, hmid, TTOK, 768, 192, 1);
        gemm_bf16_wmma<<<dim3(MB, 192 / GT_N), 256, 0, stream>>>(
            hmid, fc2W + (size_t)d * 192 * 768, fc2_b + d * 192, e,
            e, nullptr, TTOK, 192, 768, 0);
    }

    // ---- Reconstruction ----
    convert_kernel<<<(T * 192) / 256, 256, 0, stream>>>(e, A_w, (int)(T * 192));
    gemm_bf16_wmma<<<dim3(MB, 128 / GT_N), 256, 0, stream>>>(
        A_w, recW, nullptr, nullptr, rec_t, nullptr, TTOK, 128, 192, 0);
    scatter_rec_kernel<<<(T * 128) / 256, 256, 0, stream>>>(rec_t, rec_b, (float*)d_out);
}
